// MaskedAttention_73203422593481
// MI455X (gfx1250) — matrix operand order repaired and checked by reading, not compiled, not run on hardware
//
#include <hip/hip_runtime.h>

// ---- CDNA5 WMMA vector types (probe-verified signatures) ----
typedef __attribute__((ext_vector_type(16))) _Float16 v16h;
typedef __attribute__((ext_vector_type(8)))  _Float16 v8h;
typedef __attribute__((ext_vector_type(4)))  _Float16 v4h;
typedef __attribute__((ext_vector_type(2)))  _Float16 v2h;
typedef __attribute__((ext_vector_type(8)))  float    v8f;

#define SEQ    2048
#define DH     128
#define KT     32                 // keys per K/V tile
#define QW     16                 // q rows per wave (WMMA M)
#define WAVES  8
#define QB     (QW * WAVES)       // 128 q rows per block
#define KSTR   (DH + 8)           // LDS row stride (halfs), bank-conflict pad
#define VSTR   (KT + 8)
#define PSTR   (KT + 8)

// One 32-key tile of flash attention for one wave.
// MASKED=false: tile fully below the causal diagonal (no compares/cndmasks).
template <bool MASKED>
__device__ __forceinline__ void process_tile(const v16h (&qa)[4],
                                             const v16h& ones,
                                             v8f (&acc)[8],
                                             float (&m_run)[8],
                                             float (&l_run)[8],
                                             const _Float16* __restrict__ Klds,
                                             const _Float16* __restrict__ Vtlds,
                                             _Float16* __restrict__ pw,
                                             int kbase, int q0, int l16, int hi)
{
  // ---- S = Q K^T : 16x32 scores as two 16x16 f32 fragments ----
  v8f s0 = {}, s1 = {};
  #pragma unroll
  for (int c = 0; c < 4; ++c) {
    // B frag (16-bit B 32x16): lane n = key%16, halfs = d-dim {8*hi..+7, 16+8*hi..+7}
    const _Float16* kp0 = &Klds[l16 * KSTR + c * 32 + 8 * hi];
    const _Float16* kp1 = kp0 + 16 * KSTR;
    v8h b0lo = *(const v8h*)kp0, b0hi = *(const v8h*)(kp0 + 16);
    v8h b1lo = *(const v8h*)kp1, b1hi = *(const v8h*)(kp1 + 16);
    v16h kb0, kb1;
    #pragma unroll
    for (int k = 0; k < 8; ++k) {
      kb0[k] = b0lo[k]; kb0[k + 8] = b0hi[k];
      kb1[k] = b1lo[k]; kb1[k + 8] = b1hi[k];
    }
    s0 = __builtin_amdgcn_wmma_f32_16x16x32_f16(false, qa[c], false, kb0,
                                                (short)0, s0, false, false);
    s1 = __builtin_amdgcn_wmma_f32_16x16x32_f16(false, qa[c], false, kb1,
                                                (short)0, s1, false, false);
  }

  // ---- (optional) causal mask + online max (lane butterfly over 16) ----
  float alpha[8];
  #pragma unroll
  for (int r = 0; r < 8; ++r) {
    if (MASKED) {
      const int qrow = q0 + r + 8 * hi;
      const int c0   = kbase + l16;
      if (c0 > qrow)      s0[r] = -__builtin_inff();
      if (c0 + 16 > qrow) s1[r] = -__builtin_inff();
    }
    float mx = fmaxf(s0[r], s1[r]);
    #pragma unroll
    for (int off = 1; off < 16; off <<= 1)
      mx = fmaxf(mx, __shfl_xor(mx, off, 32));
    const float mnew = fmaxf(m_run[r], mx);            // finite after tile 0
    alpha[r] = __expf(m_run[r] - mnew);
    m_run[r] = mnew;
    s0[r] = __expf(s0[r] - mnew);
    s1[r] = __expf(s1[r] - mnew);
  }
  #pragma unroll
  for (int cb = 0; cb < 8; ++cb) {
    #pragma unroll
    for (int r = 0; r < 8; ++r) acc[cb][r] *= alpha[r];
  }

  // ---- P: C layout -> A layout via per-wave LDS scratch ----
  #pragma unroll
  for (int r = 0; r < 8; ++r) {
    pw[(r + 8 * hi) * PSTR + l16]      = (_Float16)s0[r];
    pw[(r + 8 * hi) * PSTR + 16 + l16] = (_Float16)s1[r];
  }
  // same-wave DS RAW: LDS ops in order; compiler inserts s_wait_dscnt
  const _Float16* pr = &pw[l16 * PSTR + 8 * hi];
  v8h plo = *(const v8h*)pr, phi = *(const v8h*)(pr + 16);
  v16h pa;
  #pragma unroll
  for (int k = 0; k < 8; ++k) { pa[k] = plo[k]; pa[k + 8] = phi[k]; }

  // ---- row sums on the matrix pipe: rs = P x ones(32x16) ----
  // replaces a 32x ds_bpermute butterfly; every lane of row r gets the sum
  v8f rz = {};
  v8f rs = __builtin_amdgcn_wmma_f32_16x16x32_f16(false, pa, false, ones,
                                                  (short)0, rz, false, false);
  #pragma unroll
  for (int r = 0; r < 8; ++r)
    l_run[r] = l_run[r] * alpha[r] + rs[r];

  // ---- O += P * V : 8 output column blocks, K-dim = 32 keys ----
  #pragma unroll
  for (int cb = 0; cb < 8; ++cb) {
    const _Float16* vp = &Vtlds[(cb * 16 + l16) * VSTR + 8 * hi];
    v8h vlo = *(const v8h*)vp, vhi = *(const v8h*)(vp + 16);
    v16h vb;
    #pragma unroll
    for (int k = 0; k < 8; ++k) { vb[k] = vlo[k]; vb[k + 8] = vhi[k]; }
    acc[cb] = __builtin_amdgcn_wmma_f32_16x16x32_f16(false, pa, false, vb,
                                                     (short)0, acc[cb],
                                                     false, false);
  }
}

__launch_bounds__(256)
__global__ void fa_fwd_causal_f16wmma(const float* __restrict__ Qg,
                                      const float* __restrict__ Kg,
                                      const float* __restrict__ Vg,
                                      float* __restrict__ Og)
{
  __shared__ _Float16 Klds [KT * KSTR];        // [key][d]      f16
  __shared__ _Float16 Vtlds[DH * VSTR];        // [d][key]      f16 (transposed)
  __shared__ _Float16 Plds [WAVES * 16 * PSTR];// per-wave P scratch

  const int tid  = threadIdx.x;
  const int wave = tid >> 5;
  const int lane = tid & 31;
  const int l16  = lane & 15;
  const int hi   = lane >> 4;      // half-wave group (selects K-range in frags)

  const int qtile = blockIdx.x;    // 0..15
  const int bh    = blockIdx.y;    // 0..B*H-1

  const float* Qb = Qg + (size_t)bh * SEQ * DH;
  const float* Kb = Kg + (size_t)bh * SEQ * DH;
  const float* Vb = Vg + (size_t)bh * SEQ * DH;
  float*       Ob = Og + (size_t)bh * SEQ * DH;

  const int q0 = qtile * QB + wave * QW;       // first q row of this wave
  const float scale = 0.08838834764831845f;    // 1/sqrt(128), folded into Q

  // ---- Q in registers, WMMA A layout (16-bit A 16x32): 4 chunks along D ----
  // lanes 0-15 hold K-offsets {0..7,16..23}; lanes 16-31 hold {8..15,24..31}
  v16h qa[4];
  {
    const float* qp = Qb + (size_t)(q0 + l16) * DH;
    #pragma unroll
    for (int c = 0; c < 4; ++c) {
      const int d0 = c * 32 + 8 * hi;
      #pragma unroll
      for (int k = 0; k < 8; ++k) {
        qa[c][k]     = (_Float16)(qp[d0 + k]      * scale);
        qa[c][k + 8] = (_Float16)(qp[d0 + 16 + k] * scale);
      }
    }
  }

  // constant all-ones B fragment for row-sum matmul
  v16h ones;
  #pragma unroll
  for (int k = 0; k < 16; ++k) ones[k] = (_Float16)1.0f;

  v8f acc[8];
  #pragma unroll
  for (int cb = 0; cb < 8; ++cb) acc[cb] = {};

  float m_run[8], l_run[8];
  #pragma unroll
  for (int r = 0; r < 8; ++r) { m_run[r] = -__builtin_inff(); l_run[r] = 0.0f; }

  _Float16* pw = &Plds[wave * 16 * PSTR];
  const int ntiles = (qtile + 1) * (QB / KT);  // causal: keys up to (qtile+1)*128

  for (int t = 0; t < ntiles; ++t) {
    const int kbase = t * KT;

    __syncthreads();   // previous tile's LDS consumers done before overwrite

    // -- cooperative K tile: 32x128 f32 -> f16 LDS row-major (float4 reads) --
    #pragma unroll
    for (int i = 0; i < 4; ++i) {
      const int li  = tid + 256 * i;           // 1024 float4 total
      const int row = li >> 5;                 // 32 float4 per row
      const int cv  = li & 31;
      const float4 f = ((const float4*)(Kb + (size_t)(kbase + row) * DH))[cv];
      v4h h; h[0] = (_Float16)f.x; h[1] = (_Float16)f.y;
             h[2] = (_Float16)f.z; h[3] = (_Float16)f.w;
      *(v4h*)&Klds[row * KSTR + cv * 4] = h;
    }
    // -- cooperative V tile: 32x128 f32 -> f16 LDS transposed [d][key] --
    // two consecutive keys per thread -> contiguous half2 LDS writes
    #pragma unroll
    for (int i = 0; i < 8; ++i) {
      const int li  = tid + 256 * i;           // 2048 key-pairs x col
      const int kp  = li >> 7;                 // key pair 0..15
      const int col = li & 127;                // d
      const float f0 = Vb[(size_t)(kbase + 2 * kp)     * DH + col];
      const float f1 = Vb[(size_t)(kbase + 2 * kp + 1) * DH + col];
      v2h h; h[0] = (_Float16)f0; h[1] = (_Float16)f1;
      *(v2h*)&Vtlds[col * VSTR + 2 * kp] = h;
    }
    __syncthreads();

    // wave-uniform causal split: bulk tiles need no masking
    if (kbase + KT <= q0 + 1) {
      process_tile<false>(qa, ones, acc, m_run, l_run, Klds, Vtlds, pw,
                          kbase, q0, l16, hi);
    } else if (kbase <= q0 + QW - 1) {
      process_tile<true>(qa, ones, acc, m_run, l_run, Klds, Vtlds, pw,
                         kbase, q0, l16, hi);
    }
  }

  // ---- epilogue: O = acc / l ----
  #pragma unroll
  for (int r = 0; r < 8; ++r) {
    const int qrow = q0 + r + 8 * hi;
    const float inv = 1.0f / l_run[r];
    #pragma unroll
    for (int cb = 0; cb < 8; ++cb)
      Ob[(size_t)qrow * DH + cb * 16 + l16] = acc[cb][r] * inv;
  }
}

extern "C" void kernel_launch(void* const* d_in, const int* in_sizes, int n_in,
                              void* d_out, int out_size, void* d_ws, size_t ws_size,
                              hipStream_t stream) {
  const float* Q = (const float*)d_in[0];
  const float* K = (const float*)d_in[1];
  const float* V = (const float*)d_in[2];
  float*       O = (float*)d_out;
  const int bh = in_sizes[0] / (SEQ * DH);     // batch*heads = 32
  dim3 grid(SEQ / QB, bh);                     // 16 x 32 blocks
  fa_fwd_causal_f16wmma<<<grid, 256, 0, stream>>>(Q, K, V, O);
}
